// GraphMixKNN_63299228008825
// MI455X (gfx1250) — matrix operand order repaired and checked by reading, not compiled, not run on hardware
//
#include <hip/hip_runtime.h>
#include <hip/hip_bf16.h>

// ---------------------------------------------------------------------------
// GraphMixKNN on MI455X (gfx1250, wave32).
//
// Roofline: ~2.25 GFLOP; mandatory HBM traffic ~38 MB (~1.6 us @ 23.3 TB/s).
// Z (12.8 MB) is L2-resident (192 MB L2) -> the K=32 random gather is
// L2-bound. GEMMs run on v_wmma_f32_16x16x32_bf16 (fp32 accumulate).
//
// Fragment feeds are layout-engineered so every WMMA operand is exactly
// two ds_load_b128 per lane:
//  - A: row-major cat tile, row pitch 272 B (68 banks = stagger 4/row,
//       conflict-free across the 16 rows); lane run K=kAbase+0..7 / +16..23.
//  - B: weights stored COLUMN-major in LDS; lane run K=kBbase+0..7 / +16..23, each 8 are
//       contiguous. Column pitch 272 B (144 B for W_gate2), 16 B aligned.
// ---------------------------------------------------------------------------

typedef __attribute__((ext_vector_type(16))) __bf16 v16bf;
typedef __attribute__((ext_vector_type(8)))  __bf16 v8bf;
typedef __attribute__((ext_vector_type(2)))  __bf16 v2bf;
typedef __attribute__((ext_vector_type(8)))  float  v8f;

#define KNBR 32
#define H    64
#define H2   128
#define GH   64

#define WAVES 4
#define TILE  16                         // nodes per wave
#define NODES_PER_BLOCK (WAVES * TILE)   // 64
#define CAT_STRIDE 136                   // bf16/row: 128 + 8 pad -> 272 B (mult of 16)
#define KSTRIDE    136                   // bf16/col for W_gate/W_msg (K=128 + 8 pad)
#define K2STRIDE   72                    // bf16/col for W_gate2     (K=64 + 8 pad)

__global__ __launch_bounds__(WAVES * 32)
void graphmix_kernel(const float* __restrict__ Z,
                     const int*   __restrict__ nbr_idx,
                     const float* __restrict__ nbr_w,
                     const float* __restrict__ vmask,
                     const float* __restrict__ W_gate,
                     const float* __restrict__ b_gate,
                     const float* __restrict__ W_gate2,
                     const float* __restrict__ b_gate2,
                     const float* __restrict__ W_msg,
                     const float* __restrict__ b_msg,
                     float* __restrict__ out,
                     int N)
{
    // ---- LDS (62.2 KB static) ----
    alignas(16) __shared__ __bf16 sWg[GH * KSTRIDE];    // 17408 B  W_gate^T  [col][K]
    alignas(16) __shared__ __bf16 sWm[GH * KSTRIDE];    // 17408 B  W_msg^T   [col][K]
    alignas(16) __shared__ __bf16 sW2[H  * K2STRIDE];   //  9216 B  W_gate2^T [col][K]
    alignas(16) __shared__ __bf16 sCat[WAVES][TILE * CAT_STRIDE];  // 17408 B
    __shared__ float sBg[GH], sB2[H], sBm[H];

    const int tid  = threadIdx.x;
    const int lane = tid & 31;
    const int wid  = tid >> 5;
    const int nthr = WAVES * 32;

    // ---- stage weights fp32 -> bf16, transposed to column-major ----
    for (int i = tid; i < H2 * GH; i += nthr) {
        int k = i >> 6, c = i & 63;                 // global row-major [K][col]
        sWg[c * KSTRIDE + k] = (__bf16)W_gate[i];
        sWm[c * KSTRIDE + k] = (__bf16)W_msg[i];
    }
    for (int i = tid; i < GH * H; i += nthr) {
        int k = i >> 6, c = i & 63;
        sW2[c * K2STRIDE + k] = (__bf16)W_gate2[i];
    }
    if (tid < GH) sBg[tid] = b_gate[tid];
    if (tid < H)  { sB2[tid] = b_gate2[tid]; sBm[tid] = b_msg[tid]; }

    const int tileBase = blockIdx.x * NODES_PER_BLOCK + wid * TILE;

    // ================= Phase 1: gather + weighted mean =================
    // Lane owns 2 features (coalesced 256 B row reads per neighbor);
    // w/idx loaded lane-per-k once, broadcast via shfl.
    for (int nl = 0; nl < TILE; ++nl) {
        int n   = tileBase + nl;
        int ncl = (n < N) ? n : (N - 1);

        float wraw = nbr_w[(size_t)ncl * KNBR + lane];
        int   iraw = nbr_idx[(size_t)ncl * KNBR + lane];
        // nan_to_num(nan/±inf->0) then clip(>=0)  ==  (finite && >0) ? w : 0
        float wk = (__builtin_isfinite(wraw) && wraw > 0.0f) ? wraw : 0.0f;

        // Warm L2/L0 with all 32 neighbor rows (global_prefetch_b8).
        __builtin_prefetch(&Z[(size_t)iraw * H], 0, 0);

        float wsum = wk;
        #pragma unroll
        for (int off = 16; off; off >>= 1) wsum += __shfl_xor(wsum, off);

        float acc0 = 0.0f, acc1 = 0.0f;
        for (int k = 0; k < KNBR; ++k) {
            float wkk = __shfl(wk, k);
            int   jj  = __shfl(iraw, k);
            float c   = wkk * vmask[jj];            // sender mask folded in
            const float* zr = &Z[(size_t)jj * H + 2 * lane];
            acc0 += c * zr[0];
            acc1 += c * zr[1];
        }
        float inv = 1.0f / (wsum + 1e-6f);

        const float* zs = &Z[(size_t)ncl * H + 2 * lane];
        __bf16* row = &sCat[wid][nl * CAT_STRIDE];
        *(v2bf*)&row[2 * lane]     = (v2bf){(__bf16)zs[0], (__bf16)zs[1]};       // cat[:,0:64]=Z
        *(v2bf*)&row[H + 2 * lane] = (v2bf){(__bf16)(acc0 * inv),
                                            (__bf16)(acc1 * inv)};               // cat[:,64:128]=Z_nb
    }

    __syncthreads();   // weights visible to all waves; cat tiles are wave-private

    // ================= Phase 2: WMMA GEMMs =================
    // ISA §7.12.2 16-bit layouts; both operands load as 2x ds_load_b128/lane.
    const int mcol   = lane & 15;
    const int kAbase = (lane < 16) ? 0 : 8;
    const int kBbase = (lane < 16) ? 0 : 8;

    const __bf16* arow = &sCat[wid][mcol * CAT_STRIDE];

    auto loadA = [&](int kc) -> v16bf {
        const __bf16* p = arow + kc * 32 + kAbase;            // 16 B aligned
        v8bf lo = *(const v8bf*)(p);                          // K = kAbase+0..7
        v8bf hi = *(const v8bf*)(p + 16);                     // K = kAbase+16..23
        return __builtin_shufflevector(lo, hi, 0,1,2,3,4,5,6,7,8,9,10,11,12,13,14,15);
    };
    auto loadB = [&](const __bf16* Wcol, int kstride, int kc, int nc) -> v16bf {
        const __bf16* p = Wcol + (nc * 16 + mcol) * kstride + kc * 32 + kBbase; // 16 B aligned
        v8bf lo = *(const v8bf*)(p);                          // K = kBbase+0..7
        v8bf hi = *(const v8bf*)(p + 16);                     // K = kBbase+16..23
        return __builtin_shufflevector(lo, hi, 0,1,2,3,4,5,6,7,8,9,10,11,12,13,14,15);
    };

    // GEMM1 (gate hidden) + GEMM-msg (delta) share A fragments.
    v8f hacc[4], dacc[4];
    #pragma unroll
    for (int nc = 0; nc < 4; ++nc) {
        float bg = sBg[nc * 16 + mcol];
        float bm = sBm[nc * 16 + mcol];
        v8f hc, dc;
        #pragma unroll
        for (int r = 0; r < 8; ++r) { hc[r] = bg; dc[r] = bm; }
        #pragma unroll
        for (int kc = 0; kc < 4; ++kc) {
            v16bf a  = loadA(kc);
            v16bf bG = loadB(sWg, KSTRIDE, kc, nc);
            hc = __builtin_amdgcn_wmma_f32_16x16x32_bf16(false, a, false, bG,
                                                         (short)0, hc, false, false);
            v16bf bM = loadB(sWm, KSTRIDE, kc, nc);
            dc = __builtin_amdgcn_wmma_f32_16x16x32_bf16(false, a, false, bM,
                                                         (short)0, dc, false, false);
        }
        #pragma unroll
        for (int r = 0; r < 8; ++r) {
            hc[r] = fmaxf(hc[r], 0.0f);   // relu(hidden)
            dc[r] = fmaxf(dc[r], 0.0f);   // relu(delta)
        }
        hacc[nc] = hc; dacc[nc] = dc;
    }

    // Re-stage relu(hidden) [16][64] bf16 over the (now dead) cat tile,
    // from the C/D register layout; close same-wave LDS RAW explicitly.
    #pragma unroll
    for (int nc = 0; nc < 4; ++nc) {
        int col = nc * 16 + mcol;
        #pragma unroll
        for (int r = 0; r < 8; ++r) {
            int M = (lane < 16) ? r : (r + 8);
            sCat[wid][M * CAT_STRIDE + col] = (__bf16)hacc[nc][r];
        }
    }
    asm volatile("s_wait_dscnt 0" ::: "memory");

    // GEMM2: hidden[16,64] @ W_gate2[64,64] -> sigmoid -> fused output.
    #pragma unroll
    for (int nc = 0; nc < 4; ++nc) {
        float b2 = sB2[nc * 16 + mcol];
        v8f gc;
        #pragma unroll
        for (int r = 0; r < 8; ++r) gc[r] = b2;
        #pragma unroll
        for (int kc = 0; kc < 2; ++kc) {
            v16bf a = loadA(kc);
            v16bf b = loadB(sW2, K2STRIDE, kc, nc);
            gc = __builtin_amdgcn_wmma_f32_16x16x32_bf16(false, a, false, b,
                                                         (short)0, gc, false, false);
        }
        int col = nc * 16 + mcol;
        #pragma unroll
        for (int r = 0; r < 8; ++r) {
            int M = (lane < 16) ? r : (r + 8);
            int n = tileBase + M;
            if (n < N) {
                float g = 1.0f / (1.0f + __expf(-gc[r]));
                out[(size_t)n * H + col] = Z[(size_t)n * H + col] + g * dacc[nc][r];
            }
        }
    }
}

extern "C" void kernel_launch(void* const* d_in, const int* in_sizes, int n_in,
                              void* d_out, int out_size, void* d_ws, size_t ws_size,
                              hipStream_t stream) {
    const float* Z       = (const float*)d_in[0];
    const int*   nbr_idx = (const int*)  d_in[1];
    const float* nbr_w   = (const float*)d_in[2];
    const float* vmask   = (const float*)d_in[3];
    const float* W_gate  = (const float*)d_in[4];
    const float* b_gate  = (const float*)d_in[5];
    const float* W_gate2 = (const float*)d_in[6];
    const float* b_gate2 = (const float*)d_in[7];
    const float* W_msg   = (const float*)d_in[8];
    const float* b_msg   = (const float*)d_in[9];
    float*       out     = (float*)d_out;

    const int N = in_sizes[0] / H;   // Z is [N, 64]
    const int blocks = (N + NODES_PER_BLOCK - 1) / NODES_PER_BLOCK;
    graphmix_kernel<<<blocks, WAVES * 32, 0, stream>>>(
        Z, nbr_idx, nbr_w, vmask, W_gate, b_gate, W_gate2, b_gate2,
        W_msg, b_msg, out, N);
}
